// MultiHeadedAttention_82944408420453
// MI455X (gfx1250) — matrix operand order repaired and checked by reading, not compiled, not run on hardware
//
#include <hip/hip_runtime.h>

typedef __bf16 bf16;
typedef __bf16 bf16x4  __attribute__((ext_vector_type(4)));
typedef __bf16 bf16x8  __attribute__((ext_vector_type(8)));
typedef __bf16 bf16x16 __attribute__((ext_vector_type(16)));
typedef float  f32x8   __attribute__((ext_vector_type(8)));

#define B_  4
#define S_  1024
#define E_  768
#define H_  12
#define HD_ 64
#define M_  (B_*S_)   // 4096

// A-fragment (16-bit, 16x32): lane group g holds K runs [g*8, g*8+8) and [g*8+16, g*8+24)
__device__ __forceinline__ bf16x16 ld_frag_rowpair(const bf16* base) {
  bf16x16 r;
  ((bf16x8*)&r)[0] = *(const bf16x8*)(base);        // ds_load_b128
  ((bf16x8*)&r)[1] = *(const bf16x8*)(base + 16);   // ds_load_b128
  return r;
}
// B-fragment from a [n][k] tile: lane group g holds K runs [g*8, g*8+8) and [g*8+16, g*8+24)
__device__ __forceinline__ bf16x16 ld_frag_contig(const bf16* base) {
  bf16x16 r;
  ((bf16x8*)&r)[0] = *(const bf16x8*)(base);
  ((bf16x8*)&r)[1] = *(const bf16x8*)(base + 16);
  return r;
}
// wave-local LDS write->read ordering (same-wave DS ops are in-order per ISA)
__device__ __forceinline__ void wave_lds_fence() {
  __builtin_amdgcn_fence(__ATOMIC_ACQ_REL, "wavefront");
  __builtin_amdgcn_wave_barrier();
}
// Async DMA: 16B global -> LDS, tracked by ASYNCcnt (CDNA5 §10 async ops)
__device__ __forceinline__ void async_copy_b128(void* lds_dst, const void* gsrc) {
  unsigned ldsoff = (unsigned)(size_t)lds_dst;      // LDS AS pointer == byte offset
  asm volatile("global_load_async_to_lds_b128 %0, %1, off"
               :: "v"(ldsoff), "v"(gsrc) : "memory");
}
__device__ __forceinline__ void wait_asynccnt0() {
  asm volatile("s_wait_asynccnt 0x0" ::: "memory");
}

// ---------------------------------------------------------------------------
// Tiled GEMM: Y = X(M x E) @ W^T + bias.  Block tile 128x128, 8 waves (2N x 4M),
// wave tile 32x64 = 8 WMMAs per K=32 step.  Double-buffered LDS, 1 barrier/step.
// MODE 0: Y bf16 scattered to (B,H,S,HD), *= scale;  MODE 1: Y f32 (M x E)
// ---------------------------------------------------------------------------
template <int MODE>
__global__ __launch_bounds__(256)
void proj_gemm_kernel(const float* __restrict__ X, const float* __restrict__ W,
                      const float* __restrict__ bias, void* __restrict__ Y,
                      float scale)
{
  __shared__ bf16 ldsX[2][128][40];   // [buf][m][k]
  __shared__ bf16 ldsW[2][128][40];   // [buf][n][k]   (B[k][n] = W[n][k])

  const int tid  = threadIdx.x;
  const int lane = tid & 31;
  const int g    = lane >> 4;
  const int ln16 = lane & 15;
  const int wv   = tid >> 5;
  const int wm   = wv & 3;            // M subtile (32 rows)
  const int wn   = wv >> 2;           // N subtile (64 cols)
  const int mblk = blockIdx.y * 128;
  const int nblk = blockIdx.x * 128;
  const int mw_l = wm * 32;
  const int nw_l = wn * 64;

  const int sr = tid >> 3;            // staging row (4 chunks, stride 32 rows)
  const int sc = (tid & 7) * 4;       // staging col

  float4 xr[4], wr[4];
  #pragma unroll
  for (int j = 0; j < 4; ++j) {
    xr[j] = *(const float4*)&X[(size_t)(mblk + sr + j * 32) * E_ + sc];
    wr[j] = *(const float4*)&W[(size_t)(nblk + sr + j * 32) * E_ + sc];
  }

  f32x8 acc[2][4] = {};
  const int NK = E_ / 32;

  for (int i = 0; i < NK; ++i) {
    const int buf = i & 1;
    #pragma unroll
    for (int j = 0; j < 4; ++j) {
      bf16x4 px, pw;
      px[0] = (bf16)xr[j].x; px[1] = (bf16)xr[j].y; px[2] = (bf16)xr[j].z; px[3] = (bf16)xr[j].w;
      pw[0] = (bf16)wr[j].x; pw[1] = (bf16)wr[j].y; pw[2] = (bf16)wr[j].z; pw[3] = (bf16)wr[j].w;
      *(bf16x4*)&ldsX[buf][sr + j * 32][sc] = px;
      *(bf16x4*)&ldsW[buf][sr + j * 32][sc] = pw;
    }
    __syncthreads();
    if (i + 1 < NK) {                 // prefetch next tile while WMMAs run
      const int kk = (i + 1) * 32;
      #pragma unroll
      for (int j = 0; j < 4; ++j) {
        xr[j] = *(const float4*)&X[(size_t)(mblk + sr + j * 32) * E_ + kk + sc];
        wr[j] = *(const float4*)&W[(size_t)(nblk + sr + j * 32) * E_ + kk + sc];
      }
    }
    bf16x16 aF0 = ld_frag_rowpair(&ldsX[buf][mw_l + ln16][g * 8]);
    bf16x16 aF1 = ld_frag_rowpair(&ldsX[buf][mw_l + 16 + ln16][g * 8]);
    #pragma unroll
    for (int sub = 0; sub < 4; ++sub) {
      bf16x16 bF = ld_frag_contig(&ldsW[buf][nw_l + sub * 16 + ln16][g * 8]);
      acc[0][sub] = __builtin_amdgcn_wmma_f32_16x16x32_bf16(
          false, aF0, false, bF, (short)0, acc[0][sub], false, false);
      acc[1][sub] = __builtin_amdgcn_wmma_f32_16x16x32_bf16(
          false, aF1, false, bF, (short)0, acc[1][sub], false, false);
    }
  }

  // ---- epilogue (straight-line; MODE folded at compile time) ----
  float bv[4];
  int   colv[4];
  #pragma unroll
  for (int sub = 0; sub < 4; ++sub) {
    colv[sub] = nblk + nw_l + sub * 16 + ln16;
    bv[sub]   = bias[colv[sub]];
  }
  #pragma unroll
  for (int mi = 0; mi < 2; ++mi) {
    #pragma unroll
    for (int r = 0; r < 8; ++r) {
      const int row = mblk + mw_l + mi * 16 + r + 8 * g;
      if (MODE == 0) {
        const int bb = row >> 10, ss = row & (S_ - 1);
        #pragma unroll
        for (int sub = 0; sub < 4; ++sub) {
          const int hh = colv[sub] >> 6, dd = colv[sub] & (HD_ - 1);
          const float val = (acc[mi][sub][r] + bv[sub]) * scale;
          ((bf16*)Y)[(((size_t)bb * H_ + hh) * S_ + ss) * HD_ + dd] = (bf16)val;
        }
      } else {
        float* yrow = (float*)Y + (size_t)row * E_;
        #pragma unroll
        for (int sub = 0; sub < 4; ++sub)
          yrow[colv[sub]] = (acc[mi][sub][r] + bv[sub]) * scale;
      }
    }
  }
}

// ---------------------------------------------------------------------------
// Flash attention: grid (S/128, B*H), 256 threads = 8 waves x 16 queries.
// Q pre-scaled by 1/sqrt(HD).  K tiles staged by the CDNA5 async DMA path
// (GLOBAL_LOAD_ASYNC_TO_LDS_B128 + s_wait_asynccnt); V staged transposed.
// ---------------------------------------------------------------------------
__global__ __launch_bounds__(256)
void flash_attn_kernel(const bf16* __restrict__ Q, const bf16* __restrict__ K,
                       const bf16* __restrict__ V, const int* __restrict__ mask,
                       float* __restrict__ XV)
{
  __shared__ bf16 ldsK[2][32][72];    // [buf][key][d]   (async-DMA filled)
  __shared__ bf16 ldsVt[2][64][40];   // [buf][d][key]   (transposed for PV B-frags)
  __shared__ bf16 ldsP[8][16][40];    // per-wave P re-stage (C-layout -> A-layout)

  const int tid  = threadIdx.x;
  const int wv   = tid >> 5;
  const int lane = tid & 31;
  const int g    = lane >> 4;
  const int ln16 = lane & 15;

  const int bh = blockIdx.y;
  const int b  = bh / H_;
  const int h  = bh % H_;
  const int q0 = blockIdx.x * 128 + wv * 16;

  const bf16* Qb = Q + (size_t)bh * S_ * HD_;
  const bf16* Kb = K + (size_t)bh * S_ * HD_;
  const bf16* Vb = V + (size_t)bh * S_ * HD_;

  // Q fragments (16x64 = two K=32 A-frags), vector-loaded from global once
  bf16x16 aQ[2];
  #pragma unroll
  for (int c = 0; c < 2; ++c)
    aQ[c] = ld_frag_rowpair(&Qb[(size_t)(q0 + ln16) * HD_ + 32 * c + g * 8]);

  // staging: one 16B chunk of K (async) and of V (register+transpose) per thread
  const int skey = tid >> 3, sd8 = (tid & 7) * 8;

  async_copy_b128(&ldsK[0][skey][sd8], &Kb[(size_t)skey * HD_ + sd8]);   // K tile 0
  bf16x8 vreg = *(const bf16x8*)&Vb[(size_t)skey * HD_ + sd8];           // V tile 0

  f32x8 o[4] = {};
  float m_i[8], l_i[8];
  #pragma unroll
  for (int r = 0; r < 8; ++r) { m_i[r] = -__builtin_inff(); l_i[r] = 0.f; }

  const int NKT = S_ / 32;
  for (int it = 0; it < NKT; ++it) {
    const int buf = it & 1;
    const int kk  = it * 32;
    #pragma unroll
    for (int i = 0; i < 8; ++i) ldsVt[buf][sd8 + i][skey] = vreg[i];
    wait_asynccnt0();                 // own K-tile(it) DMA landed in LDS
    __syncthreads();                  // ... for every wave in the block
    if (it + 1 < NKT) {               // kick next K DMA + V prefetch (overlaps WMMA)
      async_copy_b128(&ldsK[buf ^ 1][skey][sd8],
                      &Kb[(size_t)(kk + 32 + skey) * HD_ + sd8]);
      vreg = *(const bf16x8*)&Vb[(size_t)(kk + 32 + skey) * HD_ + sd8];
    }

    // S = Q @ K^T : two 16-key tiles, HD=64 contraction = 2 WMMAs each
    f32x8 sacc[2] = {};
    #pragma unroll
    for (int nt = 0; nt < 2; ++nt) {
      #pragma unroll
      for (int c = 0; c < 2; ++c) {
        bf16x16 bK = ld_frag_contig(&ldsK[buf][nt * 16 + ln16][c * 32 + g * 8]);
        sacc[nt] = __builtin_amdgcn_wmma_f32_16x16x32_bf16(
            false, aQ[c], false, bK, (short)0, sacc[nt], false, false);
      }
    }

    // mask + online softmax (row r+8g striped across the 16 lanes of this half)
    float p0[8], p1[8], scl[8];
    #pragma unroll
    for (int r = 0; r < 8; ++r) {
      const int qrow = q0 + r + 8 * g;
      const int* mrow = mask + ((size_t)b * S_ + qrow) * S_ + kk;
      float s0 = mrow[ln16]      ? sacc[0][r] : -__builtin_inff();
      float s1 = mrow[16 + ln16] ? sacc[1][r] : -__builtin_inff();
      float rm = fmaxf(s0, s1);
      #pragma unroll
      for (int off = 1; off < 16; off <<= 1)
        rm = fmaxf(rm, __shfl_xor(rm, off, 16));
      float mnew = fmaxf(m_i[r], rm);
      float sc = (m_i[r] == -__builtin_inff()) ? 0.f : __expf(m_i[r] - mnew);
      float e0 = (s0 == -__builtin_inff()) ? 0.f : __expf(s0 - mnew);
      float e1 = (s1 == -__builtin_inff()) ? 0.f : __expf(s1 - mnew);
      float rs = e0 + e1;
      #pragma unroll
      for (int off = 1; off < 16; off <<= 1)
        rs += __shfl_xor(rs, off, 16);
      l_i[r] = l_i[r] * sc + rs;
      m_i[r] = mnew;
      p0[r] = e0; p1[r] = e1; scl[r] = sc;
    }
    #pragma unroll
    for (int sub = 0; sub < 4; ++sub)
      #pragma unroll
      for (int r = 0; r < 8; ++r)
        o[sub][r] *= scl[r];

    // re-stage P (C-layout -> A-layout); per-wave buffer, wave-local ordering only
    #pragma unroll
    for (int r = 0; r < 8; ++r) {
      ldsP[wv][r + 8 * g][ln16]      = (bf16)p0[r];
      ldsP[wv][r + 8 * g][16 + ln16] = (bf16)p1[r];
    }
    wave_lds_fence();
    bf16x16 aP = ld_frag_rowpair(&ldsP[wv][ln16][g * 8]);

    // O += P(16x32) @ V(32x64) : 4 WMMAs, B-frags contiguous from ldsVt
    #pragma unroll
    for (int sub = 0; sub < 4; ++sub) {
      bf16x16 bV = ld_frag_contig(&ldsVt[buf][sub * 16 + ln16][g * 8]);
      o[sub] = __builtin_amdgcn_wmma_f32_16x16x32_bf16(
          false, aP, false, bV, (short)0, o[sub], false, false);
    }
  }

  // finalize: /= l  (fully-masked rows: l==0 -> 0, matching NaN->0 rule)
  #pragma unroll
  for (int r = 0; r < 8; ++r) {
    const float inv = (l_i[r] > 0.f) ? (1.f / l_i[r]) : 0.f;
    const int qrow = q0 + r + 8 * g;
    float* xrow = XV + ((size_t)b * S_ + qrow) * E_ + h * HD_;
    #pragma unroll
    for (int sub = 0; sub < 4; ++sub)
      xrow[sub * 16 + ln16] = o[sub][r] * inv;
  }
}

// ---------------------------------------------------------------------------
extern "C" void kernel_launch(void* const* d_in, const int* in_sizes, int n_in,
                              void* d_out, int out_size, void* d_ws, size_t ws_size,
                              hipStream_t stream)
{
  const float* query = (const float*)d_in[0];
  const float* key   = (const float*)d_in[1];
  const float* value = (const float*)d_in[2];
  const int*   mask  = (const int*)d_in[3];
  const float* Wq = (const float*)d_in[4];  const float* bq = (const float*)d_in[5];
  const float* Wk = (const float*)d_in[6];  const float* bk = (const float*)d_in[7];
  const float* Wv = (const float*)d_in[8];  const float* bv = (const float*)d_in[9];
  const float* Wo = (const float*)d_in[10]; const float* bo = (const float*)d_in[11];

  const size_t qkv_elems = (size_t)B_ * H_ * S_ * HD_;   // 3,145,728
  bf16*  q_ws  = (bf16*)d_ws;
  bf16*  k_ws  = q_ws + qkv_elems;
  bf16*  v_ws  = k_ws + qkv_elems;
  float* xv_ws = (float*)(v_ws + qkv_elems);             // (B,S,E) f32

  dim3 gblk(E_ / 128, M_ / 128);
  // fold 1/sqrt(HD)=0.125 into Q projection
  proj_gemm_kernel<0><<<gblk, 256, 0, stream>>>(query, Wq, bq, (void*)q_ws, 0.125f);
  proj_gemm_kernel<0><<<gblk, 256, 0, stream>>>(key,   Wk, bk, (void*)k_ws, 1.0f);
  proj_gemm_kernel<0><<<gblk, 256, 0, stream>>>(value, Wv, bv, (void*)v_ws, 1.0f);

  dim3 ablk(S_ / 128, B_ * H_);
  flash_attn_kernel<<<ablk, 256, 0, stream>>>(q_ws, k_ws, v_ws, mask, xv_ws);

  proj_gemm_kernel<1><<<gblk, 256, 0, stream>>>(xv_ws, Wo, bo, d_out, 1.0f);
}
